// MyGateAttention_49031346651278
// MI455X (gfx1250) — matrix operand order repaired and checked by reading, not compiled, not run on hardware
//
#include <hip/hip_runtime.h>

// ---------------- problem constants ----------------
#define BSZ   384            // batch (== F)
#define FDIM  384            // out_features
#define HDIM  64             // hidden dim (L2-norm axis)
#define FH    (FDIM * HDIM)  // 24576 projected columns
#define KDIM  384            // GEMM-1 reduction dim (== F)

typedef _Float16 v16h __attribute__((ext_vector_type(16)));
typedef _Float16 v8h  __attribute__((ext_vector_type(8)));
typedef _Float16 v4h  __attribute__((ext_vector_type(4)));
typedef float    v8f  __attribute__((ext_vector_type(8)));

// CDNA5 async global->LDS copy (16B per lane), tracked on ASYNCcnt.
// vdst = 32-bit LDS byte offset, vaddr = 64-bit global address (GV mode).
__device__ __forceinline__ void async_copy_b128(void* lds_ptr, const void* gptr) {
    unsigned           lds = (unsigned)(uintptr_t)lds_ptr;          // low 32 bits = LDS offset
    unsigned long long ga  = (unsigned long long)(uintptr_t)gptr;
    asm volatile("global_load_async_to_lds_b128 %0, %1, off"
                 :: "v"(lds), "v"(ga) : "memory");
}
__device__ __forceinline__ void wait_async0() {
    asm volatile("s_wait_asynccnt 0x0" ::: "memory");
}

// =====================================================================
// Kernel 1: q/k projection (x @ W^T + b), L2-normalize over H=64 groups,
// store f16 to workspace.  Grid: (FH/64, BSZ/64, 2{q,k}), 256 threads.
// =====================================================================
__global__ __launch_bounds__(256)
void proj_norm_kernel(const float* __restrict__ x,
                      const float* __restrict__ Wq, const float* __restrict__ bq,
                      const float* __restrict__ Wk, const float* __restrict__ bk,
                      _Float16* __restrict__ q_ws, _Float16* __restrict__ k_ws)
{
    const int  j0  = blockIdx.x * 64;           // output-column tile (one H group)
    const int  m0  = blockIdx.y * 64;           // batch-row tile
    const bool isK = (blockIdx.z != 0);
    const float* __restrict__ W    = isK ? Wk : Wq;
    const float* __restrict__ bias = isK ? bk : bq;
    _Float16*   __restrict__ outp  = isK ? k_ws : q_ws;

    // LDS staging.  Row stride 40 halves = 80B: 16B-aligned chunks, bank stride 20.
    __shared__ _Float16 As[64][40];      // x   tile  [M=64][K=32] f16
    __shared__ _Float16 Bs[64][40];      // W   tile  [N=64][K=32] f16 (row n = W row j0+n)
    __shared__ float    Ct[64][66];      // f32 C tile for bias+norm fusion
    __shared__ float    scaleR[64];

    const int t    = threadIdx.x;        // 0..255 (8 waves)
    const int lane = t & 31;
    const int wave = t >> 5;             // 0..7
    const int sel  = (lane >> 4) & 1;    // lane half (0: lanes 0-15, 1: lanes 16-31)
    const int l16  = lane & 15;

    const int sm0 = wave >> 2;           // wave handles subtiles (sm0, sn) and (sm0+2, sn)
    const int sn  = wave & 3;

    v8f acc0 = {};
    v8f acc1 = {};

    for (int kk = 0; kk < KDIM; kk += 32) {
        // ---- cooperative load: 64x32 f32 -> f16 LDS, for A and B ----
        #pragma unroll
        for (int i = 0; i < 2; ++i) {
            const int idx = t + i * 256;             // 0..511
            const int r   = idx >> 3;
            const int c4  = (idx & 7) * 4;
            float4 va = *(const float4*)(x + (size_t)(m0 + r) * KDIM + kk + c4);
            v4h ha = { (_Float16)va.x, (_Float16)va.y, (_Float16)va.z, (_Float16)va.w };
            *(v4h*)&As[r][c4] = ha;
            float4 vb = *(const float4*)(W + (size_t)(j0 + r) * KDIM + kk + c4);
            v4h hb = { (_Float16)vb.x, (_Float16)vb.y, (_Float16)vb.z, (_Float16)vb.w };
            *(v4h*)&Bs[r][c4] = hb;
            // overlap next slab's L2->WGP fill with the WMMA phase
            if (kk + 32 < KDIM) {
                __builtin_prefetch(x + (size_t)(m0 + r) * KDIM + kk + 32 + c4, 0, 3);
                __builtin_prefetch(W + (size_t)(j0 + r) * KDIM + kk + 32 + c4, 0, 3);
            }
        }
        __syncthreads();

        // ---- fragments per ISA layout ----
        // A (16x32, 16-bit): lane row = l16; elems 0..7 -> K = sel*8+0..7,
        //                    elems 8..15 -> K = 16+sel*8+0..7.
        const int rowA0 = sm0 * 16 + l16;
        const int rowA1 = (sm0 + 2) * 16 + l16;
        v16h af0, af1, bf;
        *((v8h*)&af0)     = *(const v8h*)&As[rowA0][sel * 8];
        *((v8h*)&af0 + 1) = *(const v8h*)&As[rowA0][16 + sel * 8];
        *((v8h*)&af1)     = *(const v8h*)&As[rowA1][sel * 8];
        *((v8h*)&af1 + 1) = *(const v8h*)&As[rowA1][16 + sel * 8];
        // B (32x16, 16-bit): lane col = l16; same K order as A (sel*8+0..7, 16+sel*8+0..7).
        const int rowB = sn * 16 + l16;
        *((v8h*)&bf)      = *(const v8h*)&Bs[rowB][sel * 8];
        *((v8h*)&bf + 1)  = *(const v8h*)&Bs[rowB][16 + sel * 8];

        acc0 = __builtin_amdgcn_wmma_f32_16x16x32_f16(false, af0, false, bf,
                                                      (short)0, acc0, false, false);
        acc1 = __builtin_amdgcn_wmma_f32_16x16x32_f16(false, af1, false, bf,
                                                      (short)0, acc1, false, false);
        __syncthreads();
    }

    // ---- bias add + park C tile in LDS (D layout: elem r -> row r+8*sel, col l16) ----
    const float bv = bias[j0 + sn * 16 + l16];
    #pragma unroll
    for (int r = 0; r < 8; ++r) {
        Ct[sm0 * 16       + 8 * sel + r][sn * 16 + l16] = acc0[r] + bv;
        Ct[(sm0 + 2) * 16 + 8 * sel + r][sn * 16 + l16] = acc1[r] + bv;
    }
    __syncthreads();

    // ---- L2 norm over the 64-column H group (tile == exactly one group) ----
    if (t < 64) {
        float s = 0.f;
        #pragma unroll
        for (int c = 0; c < 64; ++c) { float v = Ct[t][c]; s += v * v; }
        scaleR[t] = 1.f / fmaxf(sqrtf(s), 1e-12f);
    }
    __syncthreads();

    // ---- scale + convert + coalesced f16 store (16 halves / thread) ----
    {
        const int r  = t >> 2;
        const int c0 = (t & 3) * 16;
        const float s = scaleR[r];
        v8h h0, h1;
        #pragma unroll
        for (int i = 0; i < 8; ++i) {
            h0[i] = (_Float16)(Ct[r][c0 + i] * s);
            h1[i] = (_Float16)(Ct[r][c0 + 8 + i] * s);
        }
        _Float16* op = outp + (size_t)(m0 + r) * FH + j0 + c0;
        *(v8h*)op       = h0;
        *(v8h*)(op + 8) = h1;
    }
}

// =====================================================================
// Kernel 2: per-batch attention.  S = Q·K^T (K=64), row softmax in
// registers, out[b,f,g] = x[f,g] * P[f,g].
// Grid: (FDIM/64, BSZ), 128 threads (4 waves; wave owns 16 f-rows x all 384 g).
// Q/K staged to LDS via CDNA5 async global->LDS copies (ASYNCcnt).
// =====================================================================
__global__ __launch_bounds__(128)
void attn_kernel(const float* __restrict__ x,
                 const _Float16* __restrict__ q_ws,
                 const _Float16* __restrict__ k_ws,
                 float* __restrict__ out)
{
    const int b  = blockIdx.y;
    const int f0 = blockIdx.x * 64;

    // Row stride 72 halves = 144B: 16B aligned, bank stride 36 (conflict-free).
    __shared__ _Float16 Qs[64][72];      //  9.2 KB
    __shared__ _Float16 Ks[384][72];     // 55.3 KB

    const int t    = threadIdx.x;        // 0..127
    const int lane = t & 31;
    const int wave = t >> 5;             // 0..3
    const int sel  = (lane >> 4) & 1;
    const int l16  = lane & 15;

    // ---- async-stage K[b] (384x64 f16) and Q strip (64x64 f16) into LDS ----
    const _Float16* kb = k_ws + (size_t)b * FH;
    #pragma unroll
    for (int i = 0; i < 24; ++i) {
        const int idx = t + i * 128;           // 0..3071 chunks of 8 halves
        const int g   = idx >> 3;
        const int c8  = (idx & 7) * 8;
        async_copy_b128(&Ks[g][c8], kb + (size_t)g * HDIM + c8);
    }
    const _Float16* qb = q_ws + (size_t)b * FH + (size_t)f0 * HDIM;
    #pragma unroll
    for (int i = 0; i < 4; ++i) {
        const int idx = t + i * 128;           // 0..511
        const int r   = idx >> 3;
        const int c8  = (idx & 7) * 8;
        async_copy_b128(&Qs[r][c8], qb + (size_t)r * HDIM + c8);
    }
    wait_async0();
    __syncthreads();

    // ---- A fragments for this wave's 16-row strip (kk = 0 and 32) ----
    const int qrow = wave * 16 + l16;
    v16h a0, a1;
    *((v8h*)&a0)     = *(const v8h*)&Qs[qrow][sel * 8];
    *((v8h*)&a0 + 1) = *(const v8h*)&Qs[qrow][16 + sel * 8];
    *((v8h*)&a1)     = *(const v8h*)&Qs[qrow][32 + sel * 8];
    *((v8h*)&a1 + 1) = *(const v8h*)&Qs[qrow][48 + sel * 8];

    // ---- 24 subtiles of 16 columns, accumulated in registers ----
    v8f acc[24];
    #pragma unroll
    for (int sn = 0; sn < 24; ++sn) {
        const int krow = sn * 16 + l16;
        v16h b0, b1;
        *((v8h*)&b0)     = *(const v8h*)&Ks[krow][sel * 8];
        *((v8h*)&b0 + 1) = *(const v8h*)&Ks[krow][16 + sel * 8];
        *((v8h*)&b1)     = *(const v8h*)&Ks[krow][32 + sel * 8];
        *((v8h*)&b1 + 1) = *(const v8h*)&Ks[krow][48 + sel * 8];
        v8f c = {};
        c = __builtin_amdgcn_wmma_f32_16x16x32_f16(false, a0, false, b0,
                                                   (short)0, c, false, false);
        c = __builtin_amdgcn_wmma_f32_16x16x32_f16(false, a1, false, b1,
                                                   (short)0, c, false, false);
        acc[sn] = c;
    }

    // ---- register softmax (tau=1) + x-gating + store ----
    // For elem r: lanes 0-15 hold row (wave*16+r), lanes 16-31 row (+8);
    // each 16-lane half spans 16 cols/subtile -> shfl_xor masks 1..8 stay in-half.
    #pragma unroll
    for (int r = 0; r < 8; ++r) {
        float m = -3.402823466e38f;
        #pragma unroll
        for (int sn = 0; sn < 24; ++sn) m = fmaxf(m, acc[sn][r]);
        #pragma unroll
        for (int off = 1; off < 16; off <<= 1) m = fmaxf(m, __shfl_xor(m, off, 32));

        float s = 0.f;
        #pragma unroll
        for (int sn = 0; sn < 24; ++sn) {
            const float p = __expf(acc[sn][r] - m);
            acc[sn][r] = p;
            s += p;
        }
        #pragma unroll
        for (int off = 1; off < 16; off <<= 1) s += __shfl_xor(s, off, 32);
        const float inv = 1.f / s;

        const int fr = f0 + wave * 16 + 8 * sel + r;     // global f row
        const float* __restrict__ xr = x + (size_t)fr * FDIM;
        float* __restrict__ orow = out + ((size_t)b * FDIM + fr) * FDIM;
        #pragma unroll
        for (int sn = 0; sn < 24; ++sn) {
            const int g = sn * 16 + l16;
            orow[g] = xr[g] * (acc[sn][r] * inv);
        }
    }
}

// =====================================================================
extern "C" void kernel_launch(void* const* d_in, const int* in_sizes, int n_in,
                              void* d_out, int out_size, void* d_ws, size_t ws_size,
                              hipStream_t stream)
{
    (void)in_sizes; (void)n_in; (void)out_size; (void)ws_size;
    const float* x  = (const float*)d_in[0];
    const float* Wq = (const float*)d_in[1];
    const float* bq = (const float*)d_in[2];
    const float* Wk = (const float*)d_in[3];
    const float* bk = (const float*)d_in[4];
    float* out = (float*)d_out;

    // workspace: normalized Q then K as f16, each BSZ*FH halves (18.9 MB each)
    _Float16* q_ws = (_Float16*)d_ws;
    _Float16* k_ws = q_ws + (size_t)BSZ * FH;

    dim3 g1(FH / 64, BSZ / 64, 2);   // (384, 6, 2)
    proj_norm_kernel<<<g1, 256, 0, stream>>>(x, Wq, bq, Wk, bk, q_ws, k_ws);

    dim3 g2(FDIM / 64, BSZ);         // (6, 384)
    attn_kernel<<<g2, 128, 0, stream>>>(x, q_ws, k_ws, out);
}
